// HiDANModel_58265526338188
// MI455X (gfx1250) — matrix operand order repaired and checked by reading, not compiled, not run on hardware
//
#include <hip/hip_runtime.h>
#include <hip/hip_bf16.h>
#include <stdint.h>

typedef __attribute__((ext_vector_type(16))) __bf16 v16bf;
typedef __attribute__((ext_vector_type(8)))  float  v8f;

#define NEGV (-1e30f)
#define NROWS 6400          // B*sl = 32*200
#define HDIM  512
#define SLEN  200
#define SFULL 201
#define VOCAB 32000
#define PADT  208           // 13 * 16 padded attention dim

#if __has_builtin(__builtin_amdgcn_global_load_async_to_lds_b128)
#define HAS_ASYNC_LDS 1
#else
#define HAS_ASYNC_LDS 0
#endif

// typed address-space pointers for the async-to-LDS builtin
typedef int v4i __attribute__((vector_size(16)));
typedef __attribute__((address_space(1))) v4i* v4i_gp;
typedef __attribute__((address_space(3))) v4i* v4i_lp;

union FragBF { uint4 u[2]; v16bf v; };

__device__ __forceinline__ unsigned short f2bf(float x) {
  unsigned int u = __float_as_uint(x);
  unsigned int r = (u + 0x7FFFu + ((u >> 16) & 1u)) >> 16;
  return (unsigned short)r;
}

__device__ __forceinline__ void wait_async0() {
#if HAS_ASYNC_LDS
#if __has_builtin(__builtin_amdgcn_s_wait_asynccnt)
  __builtin_amdgcn_s_wait_asynccnt(0);
#else
  asm volatile("s_wait_asynccnt 0" ::: "memory");
#endif
#endif
}

// ---------------- cas_len per batch ----------------
__global__ void k_caslen(const int* __restrict__ cas, int* __restrict__ clen) {
  int b = blockIdx.x;
  __shared__ int cnt;
  if (threadIdx.x == 0) cnt = 0;
  __syncthreads();
  int c = 0;
  for (int i = threadIdx.x; i < SLEN; i += blockDim.x)
    if (cas[b * SFULL + i] != 0) c++;
  atomicAdd(&cnt, c);
  __syncthreads();
  if (threadIdx.x == 0) clen[b] = cnt;
}

// ---------------- embedding / time gathers -> bf16 ----------------
__global__ void k_gather(const int* __restrict__ cas, const int* __restrict__ tii,
                         const float* __restrict__ emb, const float* __restrict__ tlam,
                         unsigned short* __restrict__ embbf, unsigned short* __restrict__ twbf) {
  int t = blockIdx.x * blockDim.x + threadIdx.x;
  if (t >= NROWS * HDIM) return;
  int r = t >> 9, h = t & (HDIM - 1);
  int b = r / SLEN, i = r - b * SLEN;
  int idx  = cas[b * SFULL + i];
  int tidx = tii[b * SFULL + i];
  embbf[t] = f2bf(emb[(size_t)idx * HDIM + h]);
  twbf[t]  = f2bf(tlam[(size_t)tidx * HDIM + h]);
}

// ---------------- pack fp32 weight (K x Ncols, row-major) into WMMA B fragments ----------------
// frag f = kt*ntiles + nt : 32 lanes x 8 dwords. lane<16 -> K=kt*32+0..15, lane>=16 -> +16..31.
// dword v holds (K=base+2v [15:0], K=base+2v+1 [31:16]) at column nt*16 + (lane&15).
__global__ void k_pack(const float* __restrict__ W, unsigned int* __restrict__ P,
                       int Ncols, int total) {
  int idx = blockIdx.x * blockDim.x + threadIdx.x;
  if (idx >= total) return;
  int f = idx >> 8;
  int w = idx & 255;
  int lane = w >> 3, v = w & 7;
  int ntiles = Ncols >> 4;
  int kt = f / ntiles, nt = f - kt * ntiles;
  int n = (nt << 4) + (lane & 15);
  int k = (kt << 5) + ((lane & 16) ? 16 : 0) + (v << 1);
  unsigned int lo = f2bf(W[(size_t)k * Ncols + n]);
  unsigned int hi = f2bf(W[(size_t)(k + 1) * Ncols + n]);
  P[idx] = lo | (hi << 16);
}

// ---------------- bf16 WMMA GEMM with LDS-staged (async) B fragments ----------------
// Block: 8 waves stacked on M -> 256 rows x 64 cols per block. Per k-step the block
// stages 4 packed B fragments (4KB) into LDS once (double-buffered, async-to-LDS when
// available) and all 8 waves consume them via ds_load_b128. Each wave computes a
// 32x64 tile (2 M-frags x 4 N-frags = 8 WMMAs per k-step).
template <int ACT, bool HASF, bool HASB, bool MASK>
__global__ __launch_bounds__(256) void k_gemm(
    const unsigned short* __restrict__ A, int lda,
    const unsigned int* __restrict__ Bp, int Ncols,
    const float* __restrict__ bias,
    float* __restrict__ Cf, int ldcf,
    unsigned short* __restrict__ Cb, int ldcb,
    int Kt, const int* __restrict__ caslen) {
  __shared__ __align__(16) unsigned int sB[2 * 1024];   // 2 x 4KB k-step buffers

  int tid  = threadIdx.x;
  int wv   = tid >> 5;
  int lane = tid & 31;
  int nGroups = Ncols >> 6;
  int bm = blockIdx.x / nGroups, bn = blockIdx.x - bm * nGroups;
  int m0 = bm * 256 + wv * 32;
  int n0 = bn << 6;
  int ntilesB = Ncols >> 4;
  int nbase   = n0 >> 4;
  int lrow    = lane & 15;
  int hiHalf  = (lane & 16) ? 1 : 0;

  auto issueB = [&](int kt, int buf) {
    const unsigned int* src = Bp + (((size_t)kt * ntilesB + nbase) << 8) + (tid << 2);
#if HAS_ASYNC_LDS
    __builtin_amdgcn_global_load_async_to_lds_b128(
        (v4i_gp)(void*)src,
        (v4i_lp)(void*)&sB[buf * 1024 + (tid << 2)],
        0, 0);
#else
    uint4 d = *(const uint4*)src;
    *(uint4*)&sB[buf * 1024 + (tid << 2)] = d;
#endif
  };

  v8f acc[2][4];
#pragma unroll
  for (int a = 0; a < 2; a++)
#pragma unroll
    for (int b = 0; b < 4; b++)
#pragma unroll
      for (int e = 0; e < 8; e++) acc[a][b][e] = 0.f;

  const unsigned short* Abase0 = A + (size_t)(m0 + lrow) * lda + (hiHalf ? 16 : 0);
  const unsigned short* Abase1 = Abase0 + (size_t)16 * lda;

  issueB(0, 0);
  wait_async0();
  __syncthreads();

  for (int kt = 0; kt < Kt; kt++) {
    int cur = kt & 1;
    if (kt + 1 < Kt) issueB(kt + 1, cur ^ 1);

    FragBF af[2];
    const unsigned short* ap0 = Abase0 + (kt << 5);
    af[0].u[0] = *(const uint4*)(ap0);
    af[0].u[1] = *(const uint4*)(ap0 + 8);
    const unsigned short* ap1 = Abase1 + (kt << 5);
    af[1].u[0] = *(const uint4*)(ap1);
    af[1].u[1] = *(const uint4*)(ap1 + 8);

    FragBF bfr[4];
#pragma unroll
    for (int b = 0; b < 4; b++) {
      const unsigned int* lp = &sB[cur * 1024 + b * 256 + (lane << 3)];
      bfr[b].u[0] = *(const uint4*)lp;
      bfr[b].u[1] = *(const uint4*)(lp + 4);
    }
#pragma unroll
    for (int a = 0; a < 2; a++)
#pragma unroll
      for (int b = 0; b < 4; b++)
        acc[a][b] = __builtin_amdgcn_wmma_f32_16x16x32_bf16(
            false, af[a].v, false, bfr[b].v, (short)0, acc[a][b], false, false);

    if (kt + 1 < Kt) {
      wait_async0();
      __syncthreads();
    }
  }

#pragma unroll
  for (int a = 0; a < 2; a++) {
#pragma unroll
    for (int v = 0; v < 8; v++) {
      int row = m0 + (a << 4) + v + (hiHalf ? 8 : 0);
      float mk = 1.f;
      if (MASK) {
        int b = row / SLEN, i = row - b * SLEN;
        mk = (i < caslen[b]) ? 1.f : 0.f;
      }
#pragma unroll
      for (int bb = 0; bb < 4; bb++) {
        int col = n0 + (bb << 4) + lrow;
        float x = acc[a][bb][v] + bias[col];
        if (ACT == 1)      x = x > 0.f ? x : (__expf(x) - 1.f);
        else if (ACT == 2) x = 1.f / (1.f + __expf(-x));
        x *= mk;
        if (HASF) Cf[(size_t)row * ldcf + col] = x;
        if (HASB) Cb[(size_t)row * ldcb + col] = f2bf(x);
      }
    }
  }
}

// ---------------- attention logits: logits[b,i,j] = sum_k head[b,i,k]*tail[b,j,k] ----------------
__global__ void k_attn_logits(const unsigned short* __restrict__ head,
                              const unsigned short* __restrict__ tail,
                              float* __restrict__ logits) {
  int wid  = (blockIdx.x * blockDim.x + threadIdx.x) >> 5;
  int lane = threadIdx.x & 31;
  if (wid >= 32 * 13 * 13) return;
  int b = wid / 169, r = wid - b * 169, it = r / 13, jt = r - it * 13;
  int lrow = lane & 15;
  int hiHalf = (lane & 16) ? 1 : 0;
  int mi = it * 16 + lrow; if (mi > SLEN - 1) mi = SLEN - 1;
  int nj = jt * 16 + lrow; if (nj > SLEN - 1) nj = SLEN - 1;
  const unsigned short* hp = head + (size_t)(b * SLEN + mi) * HDIM + (hiHalf ? 16 : 0);
  const unsigned short* tp = tail + (size_t)(b * SLEN + nj) * HDIM + (hiHalf ? 16 : 0);
  v8f acc;
#pragma unroll
  for (int e = 0; e < 8; e++) acc[e] = 0.f;
  for (int kt = 0; kt < HDIM / 32; kt++) {
    FragBF af, bf;
    af.u[0] = *(const uint4*)(hp + (kt << 5));
    af.u[1] = *(const uint4*)(hp + (kt << 5) + 8);
    bf.u[0] = *(const uint4*)(tp + (kt << 5));       // 16 contiguous K-values of one column j
    bf.u[1] = *(const uint4*)(tp + (kt << 5) + 8);
    acc = __builtin_amdgcn_wmma_f32_16x16x32_bf16(
        false, af.v, false, bf.v, (short)0, acc, false, false);
  }
#pragma unroll
  for (int v = 0; v < 8; v++) {
    int ii = it * 16 + v + (hiHalf ? 8 : 0);
    int jj = jt * 16 + lrow;
    logits[((size_t)b * PADT + ii) * PADT + jj] = acc[v];
  }
}

// ---------------- masked softmax over j (one wave per (b,i) row), in-place -> score ----------------
__global__ void k_attn_softmax(float* __restrict__ sc, const int* __restrict__ caslen) {
  int wid  = (blockIdx.x * blockDim.x + threadIdx.x) >> 5;
  int lane = threadIdx.x & 31;
  if (wid >= NROWS) return;
  int b = wid / SLEN, i = wid - b * SLEN;
  int len = caslen[b];
  float* row = sc + ((size_t)b * PADT + i) * PADT;
  float mx = -3.4e38f;
  for (int j = lane; j < SLEN; j += 32) {
    bool m = (j < i) && (j < len);
    float x = m ? row[j] : NEGV;
    mx = fmaxf(mx, x);
  }
  for (int o = 16; o > 0; o >>= 1) mx = fmaxf(mx, __shfl_xor(mx, o, 32));
  float sum = 0.f;
  for (int j = lane; j < SLEN; j += 32) {
    bool m = (j < i) && (j < len);
    float x = m ? row[j] : NEGV;
    sum += __expf(x - mx);
  }
  for (int o = 16; o > 0; o >>= 1) sum += __shfl_xor(sum, o, 32);
  float inv = 1.f / sum;
  for (int j = lane; j < SLEN; j += 32) {
    bool m = (j < i) && (j < len);
    float x = m ? row[j] : NEGV;
    row[j] = m ? __expf(x - mx) * inv : 0.f;
  }
}

// ---------------- depend[b,i,h] = sum_j score[b,i,j] * hidden[b,j,h]; bf16 copy -> Acat col 512+h ----------------
__global__ void k_depend(const float* __restrict__ sc, const float* __restrict__ hidden,
                         float* __restrict__ dep, unsigned short* __restrict__ Acat) {
  int t = blockIdx.x * blockDim.x + threadIdx.x;
  if (t >= NROWS * HDIM) return;
  int r = t >> 9, h = t & (HDIM - 1);
  int b = r / SLEN, i = r - b * SLEN;
  const float* srow = sc + ((size_t)b * PADT + i) * PADT;
  const float* hcol = hidden + (size_t)b * SLEN * HDIM + h;
  float acc = 0.f;
  for (int j = 0; j < SLEN; j++) acc += srow[j] * hcol[(size_t)j * HDIM];
  dep[t] = acc;
  Acat[(size_t)r * (2 * HDIM) + HDIM + h] = f2bf(acc);
}

// ---------------- enc = (gate*hidden + (1-gate)*depend) * mask ----------------
__global__ void k_enc(const float* __restrict__ g, const float* __restrict__ hid,
                      const float* __restrict__ dep, const int* __restrict__ caslen,
                      float* __restrict__ encf, unsigned short* __restrict__ encb) {
  int t = blockIdx.x * blockDim.x + threadIdx.x;
  if (t >= NROWS * HDIM) return;
  int r = t >> 9;
  int b = r / SLEN, i = r - b * SLEN;
  float mk = (i < caslen[b]) ? 1.f : 0.f;
  float gg = g[t];
  float e = (gg * hid[t] + (1.f - gg) * dep[t]) * mk;
  encf[t] = e;
  encb[t] = f2bf(e);
}

// ---------------- map2[r] = sum_h map1[r,h]*ti[r,h]*wm2[h] + bm2 (one wave per row) ----------------
__global__ void k_map2(const float* __restrict__ m1, const float* __restrict__ ti,
                       const float* __restrict__ wm2, const float* __restrict__ bm2,
                       float* __restrict__ map2) {
  int wid  = (blockIdx.x * blockDim.x + threadIdx.x) >> 5;
  int lane = threadIdx.x & 31;
  if (wid >= NROWS) return;
  const float* a = m1 + (size_t)wid * HDIM;
  const float* c = ti + (size_t)wid * HDIM;
  float s = 0.f;
  for (int h = lane; h < HDIM; h += 32) s += a[h] * c[h] * wm2[h];
  for (int o = 16; o > 0; o >>= 1) s += __shfl_xor(s, o, 32);
  if (lane == 0) map2[wid] = s + bm2[0];
}

// ---------------- pool = softmax_i(map2 + NEG*(1-mask)) * mask (one block per batch) ----------------
__global__ void k_pool(const float* __restrict__ map2, const int* __restrict__ caslen,
                       float* __restrict__ pool) {
  int b = blockIdx.x;
  int len = caslen[b];
  __shared__ float red[256];
  int i = threadIdx.x;
  float m = (i < SLEN && i < len) ? 1.f : 0.f;
  float x = (i < SLEN) ? (map2[b * SLEN + i] + NEGV * (1.f - m)) : -3.4e38f;
  red[i] = x; __syncthreads();
  for (int o = 128; o > 0; o >>= 1) { if (i < o) red[i] = fmaxf(red[i], red[i + o]); __syncthreads(); }
  float mx = red[0]; __syncthreads();
  float e = (i < SLEN) ? __expf(x - mx) : 0.f;
  red[i] = e; __syncthreads();
  for (int o = 128; o > 0; o >>= 1) { if (i < o) red[i] += red[i + o]; __syncthreads(); }
  float sum = red[0];
  if (i < SLEN) pool[b * SLEN + i] = e / sum * m;
}

// ---------------- out_bf = bf16(pool[r] * enc[r,h]) ----------------
__global__ void k_scale(const float* __restrict__ encf, const float* __restrict__ pool,
                        unsigned short* __restrict__ outb) {
  int t = blockIdx.x * blockDim.x + threadIdx.x;
  if (t >= NROWS * HDIM) return;
  int r = t >> 9;
  outb[t] = f2bf(pool[r] * encf[t]);
}

// =====================================================================
template <int ACT, bool HASF, bool HASB, bool MASK>
static void launch_gemm(hipStream_t stream,
                        const unsigned short* A, int lda,
                        const unsigned int* Bp, int Ncols, const float* bias,
                        float* Cf, int ldcf, unsigned short* Cb, int ldcb,
                        int Ktotal, const int* mask) {
  int blocks = (NROWS / 256) * (Ncols / 64);
  k_gemm<ACT, HASF, HASB, MASK><<<blocks, 256, 0, stream>>>(
      A, lda, Bp, Ncols, bias, Cf, ldcf, Cb, ldcb, Ktotal / 32, mask);
}

extern "C" void kernel_launch(void* const* d_in, const int* in_sizes, int n_in,
                              void* d_out, int out_size, void* d_ws, size_t ws_size,
                              hipStream_t stream) {
  const int*   cas  = (const int*)d_in[0];
  const int*   tii  = (const int*)d_in[1];
  const float* emb  = (const float*)d_in[2];
  const float* tlam = (const float*)d_in[3];
  const float* w1   = (const float*)d_in[4];
  const float* b1   = (const float*)d_in[5];
  const float* wh   = (const float*)d_in[6];
  const float* bh   = (const float*)d_in[7];
  const float* wt   = (const float*)d_in[8];
  const float* bt   = (const float*)d_in[9];
  const float* wg   = (const float*)d_in[10];
  const float* bg   = (const float*)d_in[11];
  const float* wm1  = (const float*)d_in[12];
  const float* bm1  = (const float*)d_in[13];
  const float* wti  = (const float*)d_in[14];
  const float* bti  = (const float*)d_in[15];
  const float* wm2  = (const float*)d_in[16];
  const float* bm2  = (const float*)d_in[17];
  const float* wout = (const float*)d_in[18];
  const float* bout = (const float*)d_in[19];
  float* out = (float*)d_out;

  // ---- workspace layout ----
  size_t off = 0;
  auto alloc = [&](size_t bytes) -> char* {
    char* p = (char*)d_ws + off;
    off = (off + bytes + 255) & ~(size_t)255;
    return p;
  };
  const size_t pk512 = (size_t)(512 / 32) * (512 / 16) * 256 * 4;  // 512KB
  unsigned int* w1pk   = (unsigned int*)alloc(pk512);
  unsigned int* whpk   = (unsigned int*)alloc(pk512);
  unsigned int* wtpk   = (unsigned int*)alloc(pk512);
  unsigned int* wm1pk  = (unsigned int*)alloc(pk512);
  unsigned int* wtipk  = (unsigned int*)alloc(pk512);
  unsigned int* wgpk   = (unsigned int*)alloc((size_t)(1024 / 32) * (512 / 16) * 256 * 4);
  unsigned int* woutpk = (unsigned int*)alloc((size_t)(512 / 32) * (VOCAB / 16) * 256 * 4);
  unsigned short* embbf = (unsigned short*)alloc((size_t)NROWS * HDIM * 2);
  unsigned short* twbf  = (unsigned short*)alloc((size_t)NROWS * HDIM * 2);
  unsigned short* Acat  = (unsigned short*)alloc((size_t)NROWS * 2 * HDIM * 2);
  float* hiddenf = (float*)alloc((size_t)NROWS * HDIM * 4);
  unsigned short* headbf = (unsigned short*)alloc((size_t)NROWS * HDIM * 2);
  unsigned short* tailbf = (unsigned short*)alloc((size_t)NROWS * HDIM * 2);
  float* scores = (float*)alloc((size_t)32 * PADT * PADT * 4);
  float* depf   = (float*)alloc((size_t)NROWS * HDIM * 4);
  float* gatef  = (float*)alloc((size_t)NROWS * HDIM * 4);
  float* encf   = (float*)alloc((size_t)NROWS * HDIM * 4);
  float* map2b  = (float*)alloc((size_t)NROWS * 4);
  float* poolb  = (float*)alloc((size_t)NROWS * 4);
  int*   clen   = (int*)alloc(32 * 4);
  // buffer reuse (producer fully consumed before reuse):
  float* map1f = depf;                     // depend consumed by k_enc
  float* tif   = gatef;                    // gate consumed by k_enc
  unsigned short* encbf = headbf;          // head consumed by attention
  unsigned short* outbf = tailbf;          // tail consumed by attention

  const int EW = NROWS * HDIM;             // 3,276,800 elementwise work items
  const int EWB = EW / 256;                // 12800 blocks

  // ---- preprocessing ----
  k_caslen<<<32, 256, 0, stream>>>(cas, clen);
  k_gather<<<EWB, 256, 0, stream>>>(cas, tii, emb, tlam, embbf, twbf);

  auto pack = [&](const float* W, unsigned int* P, int K, int Ncols) {
    int total = (K / 32) * (Ncols / 16) * 256;
    k_pack<<<(total + 255) / 256, 256, 0, stream>>>(W, P, Ncols, total);
  };
  pack(w1, w1pk, 512, 512);
  pack(wh, whpk, 512, 512);
  pack(wt, wtpk, 512, 512);
  pack(wg, wgpk, 1024, 512);
  pack(wm1, wm1pk, 512, 512);
  pack(wti, wtipk, 512, 512);
  pack(wout, woutpk, 512, VOCAB);

  // hidden = elu(emb@w1+b1)*mask   -> f32 + bf16 (into Acat cols [0,512))
  launch_gemm<1, true, true, true>(stream, embbf, HDIM, w1pk, HDIM, b1,
                                   hiddenf, HDIM, Acat, 2 * HDIM, HDIM, clen);
  // head / tail = hidden@w{h,t}+b  -> bf16
  launch_gemm<0, false, true, false>(stream, Acat, 2 * HDIM, whpk, HDIM, bh,
                                     nullptr, 0, headbf, HDIM, HDIM, nullptr);
  launch_gemm<0, false, true, false>(stream, Acat, 2 * HDIM, wtpk, HDIM, bt,
                                     nullptr, 0, tailbf, HDIM, HDIM, nullptr);

  // attention
  {
    int waves = 32 * 13 * 13;
    k_attn_logits<<<(waves * 32 + 255) / 256, 256, 0, stream>>>(headbf, tailbf, scores);
  }
  k_attn_softmax<<<(NROWS * 32 + 255) / 256, 256, 0, stream>>>(scores, clen);
  k_depend<<<EWB, 256, 0, stream>>>(scores, hiddenf, depf, Acat);

  // gate = sigmoid(concat(hidden,depend)@wg+bg)
  launch_gemm<2, true, false, false>(stream, Acat, 2 * HDIM, wgpk, HDIM, bg,
                                     gatef, HDIM, nullptr, 0, 2 * HDIM, nullptr);
  // enc
  k_enc<<<EWB, 256, 0, stream>>>(gatef, hiddenf, depf, clen, encf, encbf);
  // map1 = elu(enc@wm1+bm1) ; ti = elu(tw@wti+bti)
  launch_gemm<1, true, false, false>(stream, encbf, HDIM, wm1pk, HDIM, bm1,
                                     map1f, HDIM, nullptr, 0, HDIM, nullptr);
  launch_gemm<1, true, false, false>(stream, twbf, HDIM, wtipk, HDIM, bti,
                                     tif, HDIM, nullptr, 0, HDIM, nullptr);
  // map2 / pool / out scaling
  k_map2<<<(NROWS * 32 + 255) / 256, 256, 0, stream>>>(map1f, tif, wm2, bm2, map2b);
  k_pool<<<32, 256, 0, stream>>>(map2b, clen, poolb);
  k_scale<<<EWB, 256, 0, stream>>>(encf, poolb, outbf);

  // logits_out = (pool*enc) @ wout + bout  -> d_out (6400 x 32000 fp32)
  launch_gemm<0, true, false, false>(stream, outbf, HDIM, woutpk, VOCAB, bout,
                                     out, VOCAB, nullptr, 0, HDIM, nullptr);

  (void)in_sizes; (void)n_in; (void)out_size; (void)ws_size;
}
